// BatchNormSPDMean_11141145166346
// MI455X (gfx1250) — compile-verified, matrix operand order checked by reading, not run on hardware
//
#include <hip/hip_runtime.h>

// ---------------- problem / algorithm constants ----------------
#define NT          256                 // threads per block = 8 wave32
#define Nn          64                  // matrix dim
#define LDA         65                  // f32 LDS row stride (conflict-free)
#define LDH         72                  // f16 staging row stride (16B-aligned rows)
#define NMAT        4096                // 64*64
#define BATCH       8192
#define KARCHER_IT  3                   // reference N_ITER
#define NS_IT       8                   // Newton-Schulz iters per sqrtm
#define N_SQRT      3                   // inverse-scaling-squaring depth for logm
#define EXP_SQ      6                   // squarings in expm
#define EXP_ORDER   8                   // Taylor order in expm
#define CHUNK       8                   // matrices per block in log pass
#define LOG_BLOCKS  (BATCH / CHUNK)     // 1024
#define RED1_BLOCKS 256
#define RED1_CHUNK  (BATCH / RED1_BLOCKS)
#define EPB         (NMAT / NT)         // 16 elements/thread

typedef __attribute__((ext_vector_type(16))) _Float16 v16h;
typedef __attribute__((ext_vector_type(8)))  _Float16 v8h;
typedef __attribute__((ext_vector_type(8)))  float    v8f;
typedef __attribute__((ext_vector_type(4)))  unsigned v4u;
typedef __attribute__((ext_vector_type(8)))  int      v8i;
typedef __attribute__((ext_vector_type(4)))  int      v4i;

// ---------------- Tensor Data Mover: DMA one 64x64 f32 tile global->LDS ---
// Builds a D# per CDNA5 ISA ch.8: group0 = {count=1 | lds_addr | global_addr
// | type=2}; group1 = {data_size=4B, tensor 64x64, tile 64x64, dim0 stride 64}.
// Issue from ONE wave only (TENSORcnt is per-wave; TDM ignores EXEC).
__device__ __forceinline__ void tdm_load_64x64_f32(void* lds_dst, const float* gsrc) {
  const unsigned lds_addr = (unsigned)(unsigned long long)(size_t)lds_dst; // addr[31:0] = LDS byte offset
  const unsigned long long ga = (unsigned long long)(size_t)gsrc;
  v4u g0;
  g0[0] = 1u;                                               // count=1 (valid user D#)
  g0[1] = lds_addr;                                         // D#.lds_addr
  g0[2] = (unsigned)(ga & 0xFFFFFFFFull);                   // global_addr[31:0]
  g0[3] = (unsigned)((ga >> 32) & 0x01FFFFFFull)            // global_addr[56:32]
        | 0x80000000u;                                      // type=2 ("image")
  v8i g1;
  g1[0] = (int)(2u << 16);        // workgroup_mask=0, data_size=2 (4 bytes)
  g1[1] = (int)(64u << 16);       // tensor_dim0 = 64  (bits 79:48)
  g1[2] = (int)(64u << 16);       // tensor_dim1 = 64  (bits 111:80)
  g1[3] = (int)(64u << 16);       // tile_dim0  = 64  (bits 127:112)
  g1[4] = 64;                     // tile_dim1  = 64, tile_dim2 = 0
  g1[5] = 64;                     // tensor_dim0_stride = 64 (data_size units)
  g1[6] = 0;
  g1[7] = 0;
  const v4i z4 = {0, 0, 0, 0};    // groups 2/3 disabled (2D tensor)
#if defined(__clang_major__) && (__clang_major__ >= 23)
  const v8i z8 = {0, 0, 0, 0, 0, 0, 0, 0};
  __builtin_amdgcn_tensor_load_to_lds(g0, g1, z4, z4, z8, 0);
#else
  __builtin_amdgcn_tensor_load_to_lds(g0, g1, z4, z4, 0);
#endif
}

// ---------------- f32 -> packed f16 staging (one conversion per matmul) ---
// dst is a 64xLDH f16 tile; S is 64x64 (ld = lda) f32, read row-contiguous.
__device__ __forceinline__ void stage_h(_Float16* dst, const float* S, int ld) {
  for (int i = threadIdx.x; i < NMAT / 2; i += NT) {
    const int r  = i >> 5;              // 32 column-pairs per row
    const int c2 = (i & 31) << 1;
    const float x0 = S[r * ld + c2];
    const float x1 = S[r * ld + c2 + 1];
    union { _Float16 h[2]; unsigned u; } p;
    p.h[0] = (_Float16)x0; p.h[1] = (_Float16)x1;   // v_cvt_pk_f16_f32
    *(unsigned*)(dst + r * LDH + c2) = p.u;          // ds_store_b32
  }
}

// Fragment load from a staged f16 tile: two ds_load_b128 per fragment.
// Works for both A (row index) and B^T (column index) operands.
__device__ __forceinline__ v16h load_frag_h(const _Float16* S, int rc0, int k0) {
  const int lane = threadIdx.x & 31;
  const int r  = rc0 + (lane & 15);
  const int kh = (lane >> 4) << 3;                   // lanes 16..31: K+8 group
  const _Float16* p = S + r * LDH + k0 + kh;
  const v8h lo = *(const v8h*)p;                     // K = k0+kh .. +7
  const v8h hi = *(const v8h*)(p + 16);              // K = k0+16+kh .. +7
  return __builtin_shufflevector(lo, hi, 0, 1, 2, 3, 4, 5, 6, 7,
                                 8, 9, 10, 11, 12, 13, 14, 15);
}

__device__ __forceinline__ void store_cd(float* C, int ld, int r0, int c0, v8f acc) {
  const int lane = threadIdx.x & 31;
  const int c  = c0 + (lane & 15);
  const int rh = (lane >> 4) << 3;
#pragma unroll
  for (int v = 0; v < 8; ++v)
    C[(r0 + rh + v) * ld + c] = acc[v];
}

// C(64x64,ldc) = A * B^T.  NOTE: every B operand in this algorithm is either
// symmetric (so this equals A*B) or we explicitly want B^T (final congruence).
// B is staged row-contiguously -> all fragment traffic is ds_load_b128.
// 8 waves; each wave produces two 16x16 tiles; K = 2 steps of 32.
__device__ __forceinline__ void mm64t(float* C, int ldc,
                                      const float* A, int lda_,
                                      const float* B, int ldb) {
  __shared__ alignas(16) _Float16 sA[Nn * LDH];
  __shared__ alignas(16) _Float16 sB[Nn * LDH];
  __syncthreads();
  stage_h(sA, A, lda_);
  stage_h(sB, B, ldb);
  __syncthreads();
  const int w = (int)(threadIdx.x >> 5);
#pragma unroll
  for (int rep = 0; rep < 2; ++rep) {
    const int t  = w + rep * 8;
    const int r0 = (t >> 2) << 4;
    const int c0 = (t & 3) << 4;
    v8f acc = {};
#pragma unroll
    for (int kk = 0; kk < Nn; kk += 32) {
      v16h a = load_frag_h(sA, r0, kk);
      v16h b = load_frag_h(sB, c0, kk);
      acc = __builtin_amdgcn_wmma_f32_16x16x32_f16(false, a, false, b,
                                                   (short)0, acc, false, false);
    }
    store_cd(C, ldc, r0, c0, acc);
  }
  __syncthreads();
}

// ---------------- elementwise helpers on 64xLDA f32 LDS tiles -------------
__device__ __forceinline__ void ew_axpbI(float* D, const float* S, float a, float b) {
  __syncthreads();
  for (int i = threadIdx.x; i < NMAT; i += NT) {
    const int r = i >> 6, c = i & 63;
    D[r * LDA + c] = a * S[r * LDA + c] + ((r == c) ? b : 0.0f);
  }
  __syncthreads();
}
__device__ __forceinline__ void ew_scale(float* D, float a) {
  __syncthreads();
  for (int i = threadIdx.x; i < NMAT; i += NT) {
    const int r = i >> 6, c = i & 63;
    D[r * LDA + c] *= a;
  }
  __syncthreads();
}
__device__ __forceinline__ void ew_acc(float* D, const float* S, float a) {
  __syncthreads();
  for (int i = threadIdx.x; i < NMAT; i += NT) {
    const int r = i >> 6, c = i & 63;
    D[r * LDA + c] += a * S[r * LDA + c];
  }
  __syncthreads();
}
__device__ __forceinline__ void ew_setI(float* D) {
  __syncthreads();
  for (int i = threadIdx.x; i < NMAT; i += NT) {
    const int r = i >> 6, c = i & 63;
    D[r * LDA + c] = (r == c) ? 1.0f : 0.0f;
  }
  __syncthreads();
}
__device__ __forceinline__ void ew_zero(float* D) {
  __syncthreads();
  for (int i = threadIdx.x; i < NMAT; i += NT) {
    const int r = i >> 6, c = i & 63;
    D[r * LDA + c] = 0.0f;
  }
  __syncthreads();
}

// Frobenius norm of a 64xLDA LDS tile (all threads get the result).
__device__ __forceinline__ float frob_norm(const float* S) {
  __shared__ float red[NT];
  __syncthreads();
  float s = 0.0f;
  for (int i = threadIdx.x; i < NMAT; i += NT) {
    const float v = S[(i >> 6) * LDA + (i & 63)];
    s += v * v;
  }
  red[threadIdx.x] = s;
  __syncthreads();
  for (int off = NT / 2; off > 0; off >>= 1) {
    if ((int)threadIdx.x < off) red[threadIdx.x] += red[threadIdx.x + off];
    __syncthreads();
  }
  const float r = sqrtf(red[0]);
  __syncthreads();
  return r;
}

// Coupled Newton-Schulz: on entry *pY holds SPD A (64xLDA tile).
// On exit *pY = sqrtm(A), *pZ = invsqrtm(A). Rotates the 4 buffers.
// All iterates are symmetric polynomials of A -> B operands symmetric.
__device__ __forceinline__ void ns_sqrt(float** pY, float** pZ, float** pW, float** pU) {
  float *Y = *pY, *Z = *pZ, *W = *pW, *U = *pU;
  const float c  = frob_norm(Y);
  const float rc = (c > 0.0f) ? 1.0f / c : 0.0f;
  const float sc = sqrtf(c > 0.0f ? c : 1.0f);
  ew_scale(Y, rc);                        // Y0 = A / ||A||_F
  ew_setI(Z);                             // Z0 = I
  for (int it = 0; it < NS_IT; ++it) {
    mm64t(W, LDA, Z, LDA, Y, LDA);        // W = Z Y      (Y sym)
    ew_axpbI(W, W, -0.5f, 1.5f);          // W = 1.5I - 0.5 Z Y
    mm64t(U, LDA, Y, LDA, W, LDA);        // U = Y W      (W sym)
    { float* t = Y; Y = U; U = t; }
    mm64t(U, LDA, W, LDA, Z, LDA);        // U = W Z      (Z sym)
    { float* t = Z; Z = U; U = t; }
  }
  ew_scale(Y, sc);                        // sqrtm(A)
  ew_scale(Z, 1.0f / sc);                 // invsqrtm(A)
  *pY = Y; *pZ = Z; *pW = W; *pU = U;
}

// E = expm(X) via scaling-and-squaring Taylor (X destroyed; T1 scratch).
__device__ __forceinline__ void expm_ss(float* E, float* X, float* T1) {
  ew_scale(X, 1.0f / (float)(1 << EXP_SQ));
  ew_setI(E);
  for (int j = EXP_ORDER; j >= 1; --j) {
    mm64t(T1, LDA, X, LDA, E, LDA);       // T1 = X E   (E sym)
    ew_axpbI(E, T1, 1.0f / (float)j, 1.0f);
  }
  for (int s = 0; s < EXP_SQ; ++s) {
    mm64t(T1, LDA, E, LDA, E, LDA);       // T1 = E E^T = E^2 (E sym)
    ew_axpbI(E, T1, 1.0f, 0.0f);
  }
}

// ---------------- kernels -------------------------------------------------
// Stage-1 batch-mean partials: 256 blocks x 32 matrices each, fixed order.
__global__ void k_mean1(const float* __restrict__ data, float* __restrict__ part) {
  const int b0 = blockIdx.x * RED1_CHUNK;
  float acc[EPB];
#pragma unroll
  for (int j = 0; j < EPB; ++j) acc[j] = 0.0f;
  for (int m = 0; m < RED1_CHUNK; ++m) {
    const float* D = data + (size_t)(b0 + m) * NMAT;
    if (m + 1 < RED1_CHUNK) __builtin_prefetch(D + NMAT, 0, 1);
#pragma unroll
    for (int j = 0; j < EPB; ++j) acc[j] += D[j * NT + threadIdx.x];
  }
#pragma unroll
  for (int j = 0; j < EPB; ++j)
    part[(size_t)blockIdx.x * NMAT + j * NT + threadIdx.x] = acc[j];
}

// Stage-2: deterministic tree over 256 partials -> M = mean(data).
__global__ void k_mean2(const float* __restrict__ part, float* __restrict__ M) {
#pragma unroll
  for (int j = 0; j < EPB; ++j) {
    float s = 0.0f;
    for (int blk = 0; blk < RED1_BLOCKS; ++blk)
      s += part[(size_t)blk * NMAT + j * NT + threadIdx.x];
    M[j * NT + threadIdx.x] = s * (1.0f / (float)BATCH);
  }
}

// Ms = sqrtm(M), Mi = invsqrtm(M)  (single block, all WMMA matmuls).
__global__ void k_sqrtinv(const float* __restrict__ M,
                          float* __restrict__ Ms, float* __restrict__ Mi) {
  __shared__ float bY[Nn * LDA], bZ[Nn * LDA], bW[Nn * LDA], bU[Nn * LDA];
  for (int i = threadIdx.x; i < NMAT; i += NT) {
    const int r = i >> 6, c = i & 63;
    bY[r * LDA + c] = 0.5f * (M[r * Nn + c] + M[c * Nn + r]);   // symmetrize
  }
  __syncthreads();
  float *Y = bY, *Z = bZ, *W = bW, *U = bU;
  ns_sqrt(&Y, &Z, &W, &U);
  for (int i = threadIdx.x; i < NMAT; i += NT) {
    const int r = i >> 6, c = i & 63;
    Ms[r * Nn + c] = Y[r * LDA + c];
    Mi[r * Nn + c] = Z[r * LDA + c];
  }
}

// Per-batch: A = Mi D Mi; logm(A) by N_SQRT NS square roots + 4-term log
// series; accumulate over CHUNK matrices into one partial per block.
// data matrices are double-buffered into LDS by the Tensor Data Mover so the
// next fetch overlaps the ~100 matmuls of the current matrix.
__global__ void k_logaccum(const float* __restrict__ data,
                           const float* __restrict__ Mi,
                           float* __restrict__ part) {
  __shared__ float b0[Nn * LDA], b1[Nn * LDA], b2[Nn * LDA], b3[Nn * LDA];
  __shared__ float bAcc[Nn * LDA];
  __shared__ alignas(16) float bD[2][NMAT];     // TDM double buffer (dense 64x64)
  const bool wave0 = ((threadIdx.x >> 5) == 0);
  ew_zero(bAcc);
  const int base = blockIdx.x * CHUNK;
  if (wave0)
    tdm_load_64x64_f32(&bD[0][0], data + (size_t)base * NMAT);
  for (int m = 0; m < CHUNK; ++m) {
    if (m + 1 < CHUNK) {
      if (wave0) {
        tdm_load_64x64_f32(&bD[(m + 1) & 1][0], data + (size_t)(base + m + 1) * NMAT);
        __builtin_amdgcn_s_wait_tensorcnt((short)1);   // in-order: buf m resident
      }
    } else {
      if (wave0) __builtin_amdgcn_s_wait_tensorcnt((short)0);
    }
    __syncthreads();                                   // publish TDM LDS writes
    const float* D = &bD[m & 1][0];
    mm64t(b0, LDA, Mi, Nn, D,  Nn);            // b0 = Mi D   (D sym)
    mm64t(b1, LDA, b0, LDA, Mi, Nn);           // b1 = Mi D Mi (Mi sym)
    float *Y = b1, *Z = b0, *W = b2, *U = b3;
    for (int s = 0; s < N_SQRT; ++s)
      ns_sqrt(&Y, &Z, &W, &U);                 // Y = A^(1/2^N_SQRT)
    // log series around I: X = Y - I; L = X - X^2/2 + X^3/3 - X^4/4
    ew_axpbI(Z, Y, 1.0f, -1.0f);               // X  (in Z, symmetric)
    ew_axpbI(W, Z, 1.0f, 0.0f);                // L = X
    ew_axpbI(U, Z, 1.0f, 0.0f);                // P = X
    float* P = U; float* tmp = Y;              // Y buffer now free
    const float coef[3] = { -0.5f, 1.0f / 3.0f, -0.25f };
    for (int k = 0; k < 3; ++k) {
      mm64t(tmp, LDA, P, LDA, Z, LDA);         // tmp = P X  (X sym)
      { float* t = P; P = tmp; tmp = t; }
      ew_acc(W, P, coef[k]);
    }
    ew_acc(bAcc, W, (float)(1 << N_SQRT));     // logm(A) = 2^s log(A^(1/2^s))
  }
  __syncthreads();
  for (int i = threadIdx.x; i < NMAT; i += NT)
    part[(size_t)blockIdx.x * NMAT + i] = bAcc[(i >> 6) * LDA + (i & 63)];
}

// T = mean of log partials; M <- Ms expm(T) Ms.
__global__ void k_update(const float* __restrict__ part,
                         const float* __restrict__ Ms,
                         float* __restrict__ M) {
  __shared__ float bT[Nn * LDA], bE[Nn * LDA], bW[Nn * LDA];
  for (int i = threadIdx.x; i < NMAT; i += NT) {
    float s = 0.0f;
    for (int blk = 0; blk < LOG_BLOCKS; ++blk)
      s += part[(size_t)blk * NMAT + i];
    bT[(i >> 6) * LDA + (i & 63)] = s * (1.0f / (float)BATCH);
  }
  __syncthreads();
  expm_ss(bE, bT, bW);                         // E = expm(T)
  mm64t(bW, LDA, Ms, Nn, bE, LDA);             // W = Ms E   (E sym)
  mm64t(M,  Nn,  bW, LDA, Ms, Nn);             // M = W Ms   (Ms sym)
}

// G = invsqrtm(sym(M)); S = expm(sym(bias)/2); P = S G.
__global__ void k_finalprep(const float* __restrict__ M,
                            const float* __restrict__ bias,
                            float* __restrict__ P) {
  __shared__ float bY[Nn * LDA], bZ[Nn * LDA], bW[Nn * LDA], bU[Nn * LDA];
  __shared__ float bE[Nn * LDA];
  for (int i = threadIdx.x; i < NMAT; i += NT) {
    const int r = i >> 6, c = i & 63;
    bY[r * LDA + c] = 0.5f * (M[r * Nn + c] + M[c * Nn + r]);
  }
  __syncthreads();
  float *Y = bY, *Z = bZ, *W = bW, *U = bU;
  ns_sqrt(&Y, &Z, &W, &U);                     // Z = G = invsqrtm(mean)
  __syncthreads();
  for (int i = threadIdx.x; i < NMAT; i += NT) {
    const int r = i >> 6, c = i & 63;
    Y[r * LDA + c] = 0.25f * (bias[r * Nn + c] + bias[c * Nn + r]); // sym(bias)/2
  }
  __syncthreads();
  expm_ss(bE, Y, W);                           // S = expm(sym(bias)/2)
  mm64t(P, Nn, bE, LDA, Z, LDA);               // P = S G    (G sym)
}

// out_b = P data_b P^T  (streaming congruence, 2 WMMA matmuls per matrix).
// data_b is pulled into LDS by the Tensor Data Mover (one D# per block).
__global__ void k_apply(const float* __restrict__ data,
                        const float* __restrict__ P,
                        float* __restrict__ out) {
  __shared__ float bT[Nn * LDA];
  __shared__ alignas(16) float bD[NMAT];
  const size_t b = blockIdx.x;
  if ((threadIdx.x >> 5) == 0) {
    tdm_load_64x64_f32(&bD[0], data + b * NMAT);
    __builtin_amdgcn_s_wait_tensorcnt((short)0);
  }
  __syncthreads();
  mm64t(bT, LDA, P, Nn, bD, Nn);               // T = P D    (D sym)
  mm64t(out + b * NMAT, Nn, bT, LDA, P, Nn);   // out = T P^T (explicit transpose)
}

// ---------------- workspace layout (floats) -------------------------------
#define OFF_M   0
#define OFF_MS  (OFF_M  + NMAT)
#define OFF_MI  (OFF_MS + NMAT)
#define OFF_P   (OFF_MI + NMAT)
#define OFF_R1  (OFF_P  + NMAT)                          // 256*4096
#define OFF_R2  (OFF_R1 + (size_t)RED1_BLOCKS * NMAT)    // 1024*4096
// total = OFF_R2 + LOG_BLOCKS*NMAT ~= 5.26M floats ~= 21 MB

extern "C" void kernel_launch(void* const* d_in, const int* in_sizes, int n_in,
                              void* d_out, int out_size, void* d_ws, size_t ws_size,
                              hipStream_t stream) {
  (void)in_sizes; (void)n_in; (void)out_size; (void)ws_size;
  const float* data = (const float*)d_in[0];   // (8192, 64, 64) f32
  const float* bias = (const float*)d_in[1];   // (64, 64) f32
  float* out = (float*)d_out;                  // (8192, 64, 64) f32
  float* ws  = (float*)d_ws;

  float* M     = ws + OFF_M;
  float* Ms    = ws + OFF_MS;
  float* Mi    = ws + OFF_MI;
  float* P     = ws + OFF_P;
  float* part1 = ws + OFF_R1;
  float* part2 = ws + OFF_R2;

  // 1) arithmetic mean of the batch (Karcher init)
  k_mean1<<<RED1_BLOCKS, NT, 0, stream>>>(data, part1);
  k_mean2<<<1, NT, 0, stream>>>(part1, M);

  // 2) affine-invariant Karcher fixed-point iterations (data L2-resident)
  for (int it = 0; it < KARCHER_IT; ++it) {
    k_sqrtinv <<<1, NT, 0, stream>>>(M, Ms, Mi);
    k_logaccum<<<LOG_BLOCKS, NT, 0, stream>>>(data, Mi, part2);
    k_update  <<<1, NT, 0, stream>>>(part2, Ms, M);
  }

  // 3) P = expm(sym(bias)/2) * invsqrtm(mean); out = P data P^T
  k_finalprep<<<1, NT, 0, stream>>>(M, bias, P);
  k_apply    <<<BATCH, NT, 0, stream>>>(data, P, out);
}
